// MultiHeadAttention_13005160972913
// MI455X (gfx1250) — matrix operand order repaired and checked by reading, not compiled, not run on hardware
//
#include <hip/hip_runtime.h>
#include <hip/hip_bf16.h>

#define EMBED 1024
#define HEADS 16
#define HDIM 64
#define BATCH 4
#define SEQ 2048
#define MROWS (BATCH * SEQ)   // 8192
#define BHN (BATCH * HEADS)   // 64

typedef __attribute__((ext_vector_type(16))) _Float16 v16h;
typedef __attribute__((ext_vector_type(8)))  _Float16 v8h;
typedef __attribute__((ext_vector_type(4)))  _Float16 v4h;
typedef __attribute__((ext_vector_type(8)))  float    v8f;
typedef __attribute__((ext_vector_type(4)))  float    v4fv;
typedef __attribute__((ext_vector_type(4)))  int      v4i;

// ---- CDNA5 async global->LDS path (guarded; falls back to ds copy)
#if defined(__has_builtin)
#if __has_builtin(__builtin_amdgcn_global_load_async_to_lds_b128)
#define HAVE_ASYNC_LDS 1
#endif
#endif
#ifndef HAVE_ASYNC_LDS
#define HAVE_ASYNC_LDS 0
#endif

__device__ __forceinline__ void g2l_16B(_Float16* l, const _Float16* g) {
#if HAVE_ASYNC_LDS
  __builtin_amdgcn_global_load_async_to_lds_b128(
      (__attribute__((address_space(1))) v4i*)(v4i*)(void*)g,
      (__attribute__((address_space(3))) v4i*)(v4i*)(void*)l, 0, 0);
#else
  *(v8h*)l = *(const v8h*)g;
#endif
}

__device__ __forceinline__ void wait_async_copies() {
#if HAVE_ASYNC_LDS
#if __has_builtin(__builtin_amdgcn_s_wait_asynccnt)
  __builtin_amdgcn_s_wait_asynccnt(0);
#else
  asm volatile("s_wait_asynccnt 0" ::: "memory");
#endif
#endif
}

// Compiler-only scheduling fence: wave-private LDS store->load pairs are
// hardware in-order (DScnt in-order per wave); just stop reordering.
__device__ __forceinline__ void lds_fence_waveprivate() {
  asm volatile("" ::: "memory");
}

// Load a 16-half fragment as two 16B chunks at p0 / p1.
__device__ __forceinline__ v16h frag2(const _Float16* p0, const _Float16* p1) {
  v8h lo = *(const v8h*)p0;
  v8h hi = *(const v8h*)p1;
  v16h r;
#pragma unroll
  for (int i = 0; i < 8; ++i) { r[i] = lo[i]; r[i + 8] = hi[i]; }
  return r;
}

__device__ __forceinline__ v8f wmma_f16(v16h a, v16h b, v8f c) {
  return __builtin_amdgcn_wmma_f32_16x16x32_f16(false, a, false, b, (short)0, c,
                                                false, false);
}

__global__ void cvt_f32_f16(const float* __restrict__ in,
                            _Float16* __restrict__ out, int n4) {
  int i = blockIdx.x * 256 + threadIdx.x;
  if (i < n4) {
    v4fv v = *(const v4fv*)(in + (size_t)i * 4);
    v4h h;
#pragma unroll
    for (int j = 0; j < 4; ++j) h[j] = (_Float16)v[j];
    *(v4h*)(out + (size_t)i * 4) = h;
  }
}

// Tiled f16 GEMM: C[M,Nc] = A[M,K] * B[K,Nc] + bias.
// Block 256 threads / 8 waves, tile 128x128, K-step 32, double-buffered LDS.
// Wave tile 32x64 -> 2x4 WMMA frags. MODE 0: scatter Q/K/Vt f16. MODE 1: f32 out.
template <int MODE>
__global__ __launch_bounds__(256) void gemm16(
    const _Float16* __restrict__ A, const _Float16* __restrict__ B,
    const float* __restrict__ bias, float* __restrict__ outf,
    _Float16* __restrict__ q16, _Float16* __restrict__ k16,
    _Float16* __restrict__ vt16, int K, int Nc) {
  __shared__ _Float16 As[2][128 * 32];
  __shared__ _Float16 Bt[2][128 * 32];  // transposed: [col][k]

  const int tid = threadIdx.x;
  const int lane = tid & 31;
  const int wave = tid >> 5;
  const int lane16 = lane & 15;
  const int hi8  = (lane >> 4) << 3;   // A-frag chunk offset
  const int hb16 = (lane >> 4) << 4;   // unused; B frag also uses hi8
  const int wrow = wave >> 1;          // 0..3
  const int wcol = wave & 1;           // 0..1
  const int m0 = blockIdx.x * 128;
  const int n0 = blockIdx.y * 128;

  const int a_row = tid >> 1, a_col = (tid & 1) << 4;
  const int b_col = tid & 127, b_kb = (tid >> 7) << 4;

  v8f acc[2][4] = {};
  const int NSTEP = K >> 5;

  auto stage = [&](int buf, int k0) {
    const _Float16* ag = A + (size_t)(m0 + a_row) * K + k0 + a_col;
    g2l_16B(&As[buf][a_row * 32 + a_col], ag);
    g2l_16B(&As[buf][a_row * 32 + a_col + 8], ag + 8);
    __builtin_prefetch(ag + 64, 0, 0);
    const _Float16* bg = B + (size_t)(k0 + b_kb) * Nc + n0 + b_col;
    __builtin_prefetch(bg + ((size_t)Nc << 5), 0, 0);
    v8h bv0, bv1;
#pragma unroll
    for (int j = 0; j < 8; ++j) bv0[j] = bg[(size_t)j * Nc];
#pragma unroll
    for (int j = 0; j < 8; ++j) bv1[j] = bg[(size_t)(j + 8) * Nc];
    *(v8h*)&Bt[buf][b_col * 32 + b_kb] = bv0;
    *(v8h*)&Bt[buf][b_col * 32 + b_kb + 8] = bv1;
  };

  stage(0, 0);
  for (int i = 0; i < NSTEP; ++i) {
    const int buf = i & 1;
    wait_async_copies();
    __syncthreads();
    if (i + 1 < NSTEP) stage(buf ^ 1, (i + 1) << 5);

    v16h af[2];
#pragma unroll
    for (int fr = 0; fr < 2; ++fr) {
      int row = wrow * 32 + fr * 16 + lane16;
      af[fr] = frag2(&As[buf][row * 32 + hi8], &As[buf][row * 32 + hi8 + 16]);
    }
#pragma unroll
    for (int fc = 0; fc < 4; ++fc) {
      int col = wcol * 64 + fc * 16 + lane16;
      const _Float16* p = &Bt[buf][col * 32 + hi8];
      v16h bf = frag2(p, p + 16);
#pragma unroll
      for (int fr = 0; fr < 2; ++fr)
        acc[fr][fc] = wmma_f16(af[fr], bf, acc[fr][fc]);
    }
  }

  if (MODE == 1) {
#pragma unroll
    for (int fr = 0; fr < 2; ++fr)
#pragma unroll
      for (int fc = 0; fc < 4; ++fc) {
        int gc = n0 + wcol * 64 + fc * 16 + lane16;
        float bb = bias[gc];
#pragma unroll
        for (int r = 0; r < 8; ++r) {
          int grow = m0 + wrow * 32 + fr * 16 + r + hi8;
          outf[(size_t)grow * Nc + gc] = acc[fr][fc][r] + bb;
        }
      }
  } else {
    // 128-col window never straddles Q/K/V: mat and bidx are block-uniform.
    const int mat = n0 >> 10;       // 0=Q, 1=K, 2=V
    const int bidx = m0 >> 11;
    const int ccb = n0 & 1023;
    if (mat < 2) {
      _Float16* dst = (mat == 0) ? q16 : k16;
#pragma unroll
      for (int fr = 0; fr < 2; ++fr)
#pragma unroll
        for (int fc = 0; fc < 4; ++fc) {
          int cw = wcol * 64 + fc * 16 + lane16;
          float bb = bias[n0 + cw];
          int cc = ccb + cw;
          int h = cc >> 6, d = cc & 63;
          _Float16* base =
              dst + ((size_t)(bidx * HEADS + h)) * SEQ * HDIM + d;
#pragma unroll
          for (int r = 0; r < 8; ++r) {
            int n = (m0 & 2047) + wrow * 32 + fr * 16 + r + hi8;
            base[(size_t)n * HDIM] = (_Float16)(acc[fr][fc][r] + bb);
          }
        }
    } else {
#pragma unroll
      for (int fr = 0; fr < 2; ++fr)
#pragma unroll
        for (int fc = 0; fc < 4; ++fc) {
          int cw = wcol * 64 + fc * 16 + lane16;
          float bb = bias[n0 + cw];
          int cc = ccb + cw;
          int h = cc >> 6, d = cc & 63;
          _Float16* base =
              vt16 + (((size_t)(bidx * HEADS + h)) * HDIM + d) * SEQ;
#pragma unroll
          for (int r = 0; r < 8; ++r) {
            int n = (m0 & 2047) + wrow * 32 + fr * 16 + r + hi8;
            base[n] = (_Float16)(acc[fr][fc][r] + bb);
          }
        }
    }
  }
}

// Flash attention: grid (SEQ/128, B*H), 8 waves x 16 query rows.
// Key blocks of 64, K/V tiles staged in double-buffered LDS shared by block.
__global__ __launch_bounds__(256) void flash_attn(
    const _Float16* __restrict__ q16, const _Float16* __restrict__ k16,
    const _Float16* __restrict__ vt16, _Float16* __restrict__ attn16) {
  __shared__ _Float16 Ks[2][64 * 64];   // [key_local][d]
  __shared__ _Float16 Vs[2][64 * 64];   // [d][key_local]
  __shared__ _Float16 Pl[8 * 16 * 64];  // per-wave P tiles

  const int tid = threadIdx.x;
  const int lane = tid & 31;
  const int wave = tid >> 5;
  const int lane16 = lane & 15;
  const int hi8  = (lane >> 4) << 3;
  const int hb16 = (lane >> 4) << 4;
  const int bh = blockIdx.y;
  const int b = bh >> 4, h = bh & 15;
  const int q0 = blockIdx.x * 128 + wave * 16;

  _Float16* Pw = Pl + wave * 16 * 64;

  const _Float16* Qb = q16 + (size_t)bh * SEQ * HDIM;
  const _Float16* Kb = k16 + (size_t)bh * SEQ * HDIM;
  const _Float16* Vb = vt16 + (size_t)bh * HDIM * SEQ;

  const int st_row = tid >> 2;        // 0..63
  const int st_col = (tid & 3) << 4;  // 0,16,32,48

  auto stage = [&](int buf, int j0) {
    const _Float16* kg = Kb + (size_t)(j0 + st_row) * HDIM + st_col;
    g2l_16B(&Ks[buf][st_row * 64 + st_col], kg);
    g2l_16B(&Ks[buf][st_row * 64 + st_col + 8], kg + 8);
    __builtin_prefetch(kg + 64 * HDIM, 0, 0);  // 1 block ahead of staged
    const _Float16* vg = Vb + (size_t)st_row * SEQ + j0 + st_col;
    g2l_16B(&Vs[buf][st_row * 64 + st_col], vg);
    g2l_16B(&Vs[buf][st_row * 64 + st_col + 8], vg + 8);
    __builtin_prefetch(vg + 64, 0, 0);
  };

  // Q fragments (A layout), with softmax scale 1/8 folded in (exact in f16).
  v16h aQ[2];
#pragma unroll
  for (int s = 0; s < 2; ++s) {
    const _Float16* p = Qb + (size_t)(q0 + lane16) * HDIM + s * 32 + hi8;
    v16h q = frag2(p, p + 16);
#pragma unroll
    for (int e = 0; e < 16; ++e) q[e] = q[e] * (_Float16)0.125f;
    aQ[s] = q;
  }

  v8f o[4] = {};
  float mrun[8], lrun[8];
#pragma unroll
  for (int r = 0; r < 8; ++r) { mrun[r] = -3.0e38f; lrun[r] = 0.f; }

  stage(0, 0);
  for (int it = 0; it < SEQ / 64; ++it) {
    const int buf = it & 1;
    wait_async_copies();
    __syncthreads();
    if (it + 1 < SEQ / 64) stage(buf ^ 1, (it + 1) * 64);

    // S = (Q/8) * K^T for 64 keys: four 16-col C frags.
    v8f sf[4] = {};
#pragma unroll
    for (int kc = 0; kc < 4; ++kc) {
      const _Float16* kp = &Ks[buf][(kc * 16 + lane16) * 64];
#pragma unroll
      for (int s = 0; s < 2; ++s) {
        v16h bK = frag2(kp + s * 32 + hi8, kp + s * 32 + hi8 + 16);
        sf[kc] = wmma_f16(aQ[s], bK, sf[kc]);
      }
    }
    // Online softmax: rows r (lanes<16) / r+8 (lanes>=16).
#pragma unroll
    for (int r = 0; r < 8; ++r) {
      float v0 = sf[0][r], v1 = sf[1][r], v2 = sf[2][r], v3 = sf[3][r];
      float mx = fmaxf(fmaxf(v0, v1), fmaxf(v2, v3));
      mx = fmaxf(mx, __shfl_xor(mx, 1));
      mx = fmaxf(mx, __shfl_xor(mx, 2));
      mx = fmaxf(mx, __shfl_xor(mx, 4));
      mx = fmaxf(mx, __shfl_xor(mx, 8));
      float mnew = fmaxf(mrun[r], mx);
      float alpha = __expf(mrun[r] - mnew);
      float p0 = __expf(v0 - mnew), p1 = __expf(v1 - mnew);
      float p2 = __expf(v2 - mnew), p3 = __expf(v3 - mnew);
      float rs = (p0 + p1) + (p2 + p3);
      rs += __shfl_xor(rs, 1);
      rs += __shfl_xor(rs, 2);
      rs += __shfl_xor(rs, 4);
      rs += __shfl_xor(rs, 8);
      lrun[r] = lrun[r] * alpha + rs;
      mrun[r] = mnew;
#pragma unroll
      for (int f = 0; f < 4; ++f) o[f][r] *= alpha;
      int prow = r + hi8;
      Pw[prow * 64 + lane16]      = (_Float16)p0;
      Pw[prow * 64 + 16 + lane16] = (_Float16)p1;
      Pw[prow * 64 + 32 + lane16] = (_Float16)p2;
      Pw[prow * 64 + 48 + lane16] = (_Float16)p3;
    }
    lds_fence_waveprivate();  // wave-private P tile: HW DS is in-order

    v16h aP[2];
#pragma unroll
    for (int s = 0; s < 2; ++s)
      aP[s] = frag2(Pw + lane16 * 64 + s * 32 + hi8,
                    Pw + lane16 * 64 + s * 32 + hi8 + 16);
#pragma unroll
    for (int f = 0; f < 4; ++f) {
      const _Float16* vp = &Vs[buf][(f * 16 + lane16) * 64];
#pragma unroll
      for (int s = 0; s < 2; ++s) {
        v16h bV = frag2(vp + s * 32 + hi8, vp + s * 32 + hi8 + 16);
        o[f] = wmma_f16(aP[s], bV, o[f]);
      }
    }
  }

#pragma unroll
  for (int r = 0; r < 8; ++r) {
    float inv = 1.0f / lrun[r];
    int grow = q0 + r + hi8;
#pragma unroll
    for (int f = 0; f < 4; ++f) {
      int d = f * 16 + lane16;
      attn16[((size_t)(b * SEQ + grow)) * EMBED + h * HDIM + d] =
          (_Float16)(o[f][r] * inv);
    }
  }
}

extern "C" void kernel_launch(void* const* d_in, const int* in_sizes, int n_in,
                              void* d_out, int out_size, void* d_ws,
                              size_t ws_size, hipStream_t stream) {
  const float* x      = (const float*)d_in[0];
  const float* w_qkv  = (const float*)d_in[1];
  const float* b_qkv  = (const float*)d_in[2];
  const float* w_proj = (const float*)d_in[3];
  const float* b_proj = (const float*)d_in[4];
  float* out = (float*)d_out;

  _Float16* p = (_Float16*)d_ws;
  _Float16* xf16    = p; p += (size_t)MROWS * EMBED;
  _Float16* wqkv16  = p; p += (size_t)EMBED * 3 * EMBED;
  _Float16* wproj16 = p; p += (size_t)EMBED * EMBED;
  _Float16* q16     = p; p += (size_t)BHN * SEQ * HDIM;
  _Float16* k16     = p; p += (size_t)BHN * SEQ * HDIM;
  _Float16* vt16    = p; p += (size_t)BHN * SEQ * HDIM;
  _Float16* attn16  = p; p += (size_t)MROWS * EMBED;

  int n1 = MROWS * EMBED / 4;
  int n2 = EMBED * 3 * EMBED / 4;
  int n3 = EMBED * EMBED / 4;
  cvt_f32_f16<<<(n1 + 255) / 256, 256, 0, stream>>>(x, xf16, n1);
  cvt_f32_f16<<<(n2 + 255) / 256, 256, 0, stream>>>(w_qkv, wqkv16, n2);
  cvt_f32_f16<<<(n3 + 255) / 256, 256, 0, stream>>>(w_proj, wproj16, n3);

  gemm16<0><<<dim3(MROWS / 128, 3 * EMBED / 128), 256, 0, stream>>>(
      xf16, wqkv16, b_qkv, nullptr, q16, k16, vt16, EMBED, 3 * EMBED);
  flash_attn<<<dim3(SEQ / 128, BHN), 256, 0, stream>>>(q16, k16, vt16, attn16);
  gemm16<1><<<dim3(MROWS / 128, EMBED / 128), 256, 0, stream>>>(
      attn16, wproj16, b_proj, out, nullptr, nullptr, nullptr, EMBED, EMBED);
}
